// PatchAttention_48679159333044
// MI455X (gfx1250) — matrix operand order repaired and checked by reading, not compiled, not run on hardware
//
#include <hip/hip_runtime.h>
#include <hip/hip_bf16.h>

#define D_MODEL 512
#define NHEAD   8
#define HD      64
#define BATCH   2
#define SEQ     2048
#define LOG2E   1.44269504088896f
#define TRY_ASYNC 1

typedef __attribute__((ext_vector_type(16))) __bf16 v16bf;
typedef __attribute__((ext_vector_type(8)))  float  v8f;

union Frag128x2 { v16bf v; uint4 q[2]; };

__device__ inline unsigned short f2bf(float f) {
    union { float f; unsigned int u; } c; c.f = f;
    unsigned int u = c.u;
    unsigned int r = u + 0x7FFFu + ((u >> 16) & 1u);   // round-to-nearest-even
    return (unsigned short)(r >> 16);
}
__device__ inline unsigned int pack2(float a, float b) {
    return (unsigned int)f2bf(a) | ((unsigned int)f2bf(b) << 16);
}

// ---- wide fragment loaders (pre-swizzled LDS layouts, all 16B aligned) ----
// A 16x32 bf16 from row-major [m][k] tile (stride ld elems, ld*2 % 16 == 0):
// lane L: m=L&15, kb=(L>>4)*8; runs k=[kb,kb+8) and [kb+16,kb+24).
__device__ inline v16bf load_A_row(const unsigned short* base, int ld, int lane) {
    Frag128x2 f;
    const unsigned short* p = base + (lane & 15) * ld + (lane >> 4) * 8;
    f.q[0] = *(const uint4*)(p);
    f.q[1] = *(const uint4*)(p + 16);
    return f.v;
}
// B 32x16 bf16 from n-major/k-contiguous [n][k] tile (stride ld elems):
// lane L: n=L&15, kb=(L>>4)*8; runs k=[kb,kb+8) and [kb+16,kb+24) -> 2 uint4.
__device__ inline v16bf load_B_nmajor(const unsigned short* base, int ld, int lane) {
    Frag128x2 f;
    const unsigned short* p = base + (lane & 15) * ld + (lane >> 4) * 8;
    f.q[0] = *(const uint4*)(p);
    f.q[1] = *(const uint4*)(p + 16);
    return f.v;
}

__device__ inline v8f wmma_bf16(v16bf a, v16bf b, v8f c) {
    return __builtin_amdgcn_wmma_f32_16x16x32_bf16(false, a, false, b,
                                                   (short)0, c, false, false);
}

// ---- CDNA5 async memory->LDS copy (16B per lane), ASYNCcnt tracked -------
__device__ inline void async_copy_b128(const void* gsrc, void* ldst) {
#if TRY_ASYNC
    unsigned int lds = (unsigned int)(uintptr_t)ldst;    // LDS byte offset
    unsigned long long g = (unsigned long long)(uintptr_t)gsrc;
    asm volatile("global_load_async_to_lds_b128 %0, %1, off"
                 :: "v"(lds), "v"(g) : "memory");
#else
    *(uint4*)ldst = *(const uint4*)gsrc;
#endif
}
__device__ inline void async_wait_all() {
#if TRY_ASYNC
    asm volatile("s_wait_asynccnt 0" ::: "memory");
#endif
}

// ===========================================================================
// Kernel 0: one-shot fp32 -> bf16 pack (8 elements / thread, b128 I/O).
// Each element converted exactly once (GEMM tiles re-read it many times).
// ===========================================================================
__global__ __launch_bounds__(256) void pack_bf16_kernel(
        const float* __restrict__ src, unsigned short* __restrict__ dst,
        int n8) {
    int idx = blockIdx.x * 256 + threadIdx.x;
    if (idx >= n8) return;
    const float4* s = (const float4*)(src + (size_t)idx * 8);
    float4 a0 = s[0], a1 = s[1];
    uint4 u;
    u.x = pack2(a0.x, a0.y); u.y = pack2(a0.z, a0.w);
    u.z = pack2(a1.x, a1.y); u.w = pack2(a1.z, a1.w);
    *(uint4*)(dst + (size_t)idx * 8) = u;
}

// ===========================================================================
// Kernel 1: coord projection  cp[b,h,n] = sum_c coords[b,n,c]*rel_weight[h,c]
// ===========================================================================
__global__ __launch_bounds__(256) void coord_proj_kernel(
        const float* __restrict__ coords, const float* __restrict__ rel_w,
        float* __restrict__ cp) {
    int idx = blockIdx.x * 256 + threadIdx.x;
    if (idx >= BATCH * NHEAD * SEQ) return;
    int n = idx & (SEQ - 1);
    int h = (idx / SEQ) & (NHEAD - 1);
    int b = idx / (SEQ * NHEAD);
    const float* cr = coords + ((size_t)b * SEQ + n) * 3;
    const float* rw = rel_w + h * 3;
    cp[idx] = cr[0] * rw[0] + cr[1] * rw[1] + cr[2] * rw[2];
}

// ===========================================================================
// Kernel 2: QKV GEMM  C[4096,1536] = X[4096,512] @ W^T + bias, bf16 WMMA.
// Block = 128 rows x 64 cols, 8 waves (16 rows each), K-steps of 32.
// Tiles staged from pre-packed bf16 buffers via async global->LDS.
// ===========================================================================
__global__ __launch_bounds__(256) void qkv_gemm_kernel(
        const unsigned short* __restrict__ xb,   // bf16 [4096][512]
        const unsigned short* __restrict__ wb,   // bf16 [1536][512]
        const float* __restrict__ bias,
        unsigned short* __restrict__ qb, unsigned short* __restrict__ kb,
        unsigned short* __restrict__ vb) {
    __shared__ __align__(16) unsigned short Xs[128][40]; // [m][k] bf16
    __shared__ __align__(16) unsigned short Wl[64][40];  // [n][k] bf16

    const int tid  = threadIdx.x;
    const int lane = tid & 31;
    const int wv   = tid >> 5;
    const int mb   = blockIdx.y * 128;
    const int jb   = blockIdx.x * 64;

    v8f acc[4];
    #pragma unroll
    for (int t = 0; t < 4; ++t)
        #pragma unroll
        for (int i = 0; i < 8; ++i) acc[t][i] = 0.0f;

    for (int kk = 0; kk < D_MODEL; kk += 32) {
        // X tile 128x32 bf16: 2 async b128 per thread
        {
            const int r = tid >> 1, c = (tid & 1) * 16;
            const unsigned short* src = &xb[(size_t)(mb + r) * D_MODEL + kk + c];
            async_copy_b128(src,     &Xs[r][c]);
            async_copy_b128(src + 8, &Xs[r][c + 8]);
        }
        // W tile 64x32 bf16 as [n][k]: 1 async b128 per thread
        {
            const int n = tid >> 2, kc = (tid & 3) * 8;
            async_copy_b128(&wb[(size_t)(jb + n) * D_MODEL + kk + kc],
                            &Wl[n][kc]);
        }
        async_wait_all();
        __syncthreads();

        v16bf a = load_A_row(&Xs[wv * 16][0], 40, lane);
        #pragma unroll
        for (int t = 0; t < 4; ++t) {
            v16bf bfr = load_B_nmajor(&Wl[t * 16][0], 40, lane);
            acc[t] = wmma_bf16(a, bfr, acc[t]);
        }
        __syncthreads();
    }

    // epilogue: bias + scatter into head-major bf16 q/k/v
    #pragma unroll
    for (int t = 0; t < 4; ++t) {
        const int j    = jb + t * 16 + (lane & 15);
        const float bj = bias[j];
        const int part = j >> 9;
        const int rem  = j & 511;
        const int hh   = rem >> 6;
        const int hd   = rem & 63;
        unsigned short* dst = (part == 0) ? qb : ((part == 1) ? kb : vb);
        const float scale = (part == 0) ? 0.125f : 1.0f;   // 1/sqrt(64)
        #pragma unroll
        for (int i = 0; i < 8; ++i) {
            const int mg = mb + wv * 16 + (lane >> 4) * 8 + i;
            const int bi = mg >> 11;
            const int nn = mg & (SEQ - 1);
            dst[(((size_t)(bi * NHEAD + hh) * SEQ + nn) * HD) + hd] =
                f2bf((acc[t][i] + bj) * scale);
        }
    }
}

// ===========================================================================
// Kernel 3: flash attention per (b,h). 8 waves x 16 q rows; kv tiles of 32.
// S^T = K @ Q^T ; online softmax; O^T = V^T @ P^T.
// ===========================================================================
__global__ __launch_bounds__(256) void flash_attn_kernel(
        const unsigned short* __restrict__ qb,
        const unsigned short* __restrict__ kbuf,
        const unsigned short* __restrict__ vbuf,
        const float* __restrict__ cp, const float* __restrict__ mask,
        float* __restrict__ out) {
    __shared__ __align__(16) unsigned short Ks[32][40];      // [kv][hd]
    __shared__ __align__(16) unsigned short Vt[64][40];      // [hd][kv]
    __shared__ __align__(16) unsigned short Pl[8][16][40];   // [wave][q][kv]
    __shared__ float cps[32];

    const int tid  = threadIdx.x;
    const int lane = tid & 31;
    const int wv   = tid >> 5;
    const int bh   = blockIdx.y;
    const int b    = bh >> 3;
    const int h    = bh & 7;
    const int q16  = blockIdx.x * 128 + wv * 16;

    const unsigned short* qh = qb   + (size_t)bh * SEQ * HD;
    const unsigned short* kh = kbuf + (size_t)bh * SEQ * HD;
    const unsigned short* vh = vbuf + (size_t)bh * SEQ * HD;

    // preload Q^T B-fragments: per lane hd runs [kb,kb+8),[kb+16,kb+24) -> 2x b128 global
    v16bf Qf[2];
    {
        const unsigned short* qrow = qh + (size_t)(q16 + (lane & 15)) * HD;
        #pragma unroll
        for (int half = 0; half < 2; ++half) {
            Frag128x2 f;
            const unsigned short* p = qrow + half * 32 + (lane >> 4) * 8;
            f.q[0] = *(const uint4*)(p);
            f.q[1] = *(const uint4*)(p + 16);
            Qf[half] = f.v;
        }
    }

    const float cpq = cp[(size_t)bh * SEQ + q16 + (lane & 15)];
    const float* maskrow = mask + (size_t)(q16 + (lane & 15)) * SEQ;

    float mrun = -1e30f, lrun = 0.0f;
    v8f acc[4];
    #pragma unroll
    for (int c = 0; c < 4; ++c)
        #pragma unroll
        for (int i = 0; i < 8; ++i) acc[c][i] = 0.0f;

    for (int kv = 0; kv < SEQ; kv += 32) {
        __syncthreads();   // previous iteration's LDS consumers done

        // prefetch next tile's mask row (streaming operand)
        if (kv + 32 < SEQ)
            __builtin_prefetch(maskrow + kv + 32, 0, 1);

        // K tile: async global->LDS, 16B per thread (32x64 bf16 = 4KB)
        {
            const int r = tid >> 3, c = (tid & 7) * 8;
            async_copy_b128(&kh[(size_t)(kv + r) * HD + c], &Ks[r][c]);
        }
        // V tile: staged transposed Vt[hd][kv] (8 scattered b16 per thread)
        {
            const int r = tid >> 3, c = (tid & 7) * 8;
            union { uint4 q; unsigned short s[8]; } vrow;
            vrow.q = *(const uint4*)&vh[(size_t)(kv + r) * HD + c];
            #pragma unroll
            for (int j = 0; j < 8; ++j) Vt[c + j][r] = vrow.s[j];
        }
        if (tid < 32) cps[tid] = cp[(size_t)bh * SEQ + kv + tid];
        async_wait_all();
        __syncthreads();

        // scores S^T[32kv x 16q] with bias; mask loaded as float4
        float sv[16];
        #pragma unroll
        for (int rb = 0; rb < 2; ++rb) {
            v8f st;
            #pragma unroll
            for (int i = 0; i < 8; ++i) st[i] = 0.0f;
            st = wmma_bf16(load_A_row(&Ks[rb * 16][0],  40, lane), Qf[0], st);
            st = wmma_bf16(load_A_row(&Ks[rb * 16][32], 40, lane), Qf[1], st);
            const int kvl0 = rb * 16 + (lane >> 4) * 8;
            const float4* mp = (const float4*)&maskrow[kv + kvl0];
            float4 m0 = mp[0], m1 = mp[1];
            const float mk[8] = {m0.x, m0.y, m0.z, m0.w, m1.x, m1.y, m1.z, m1.w};
            #pragma unroll
            for (int i = 0; i < 8; ++i)
                sv[rb * 8 + i] = st[i] + cpq - cps[kvl0 + i] + mk[i];
        }

        // online softmax (q column lives in lanes L and L+16)
        float mloc = -1e30f;
        #pragma unroll
        for (int j = 0; j < 16; ++j) mloc = fmaxf(mloc, sv[j]);
        mloc = fmaxf(mloc, __shfl_xor(mloc, 16));
        const float mnew = fmaxf(mrun, mloc);
        const float rescale = __builtin_exp2f((mrun - mnew) * LOG2E);

        float rs = 0.0f;
        #pragma unroll
        for (int rb = 0; rb < 2; ++rb) {
            float p[8];
            #pragma unroll
            for (int i = 0; i < 8; ++i) {
                p[i] = __builtin_exp2f((sv[rb * 8 + i] - mnew) * LOG2E);
                rs += p[i];
            }
            uint4 u;
            u.x = pack2(p[0], p[1]); u.y = pack2(p[2], p[3]);
            u.z = pack2(p[4], p[5]); u.w = pack2(p[6], p[7]);
            // P^T stored [q][kv]: lane's 8 values are kv-contiguous
            *(uint4*)&Pl[wv][lane & 15][rb * 16 + (lane >> 4) * 8] = u;
        }
        rs += __shfl_xor(rs, 16);
        lrun = lrun * rescale + rs;
        mrun = mnew;
        #pragma unroll
        for (int c = 0; c < 4; ++c)
            #pragma unroll
            for (int i = 0; i < 8; ++i) acc[c][i] *= rescale;

        __syncthreads();   // P^T visible; Ks done, Vt still live

        // O^T += V^T @ P^T  (4 hd chunks of 16), B fragment reused 4x
        v16bf pv = load_B_nmajor(&Pl[wv][0][0], 40, lane);
        #pragma unroll
        for (int c = 0; c < 4; ++c) {
            v16bf av = load_A_row(&Vt[c * 16][0], 40, lane);
            acc[c] = wmma_bf16(av, pv, acc[c]);
        }
    }

    // epilogue: normalize, vectorized float4 stores
    const float inv = 1.0f / lrun;
    const int q = q16 + (lane & 15);
    float* orow = out + ((size_t)b * SEQ + q) * D_MODEL + h * HD;
    #pragma unroll
    for (int c = 0; c < 4; ++c) {
        float4 o0, o1;
        o0.x = acc[c][0] * inv; o0.y = acc[c][1] * inv;
        o0.z = acc[c][2] * inv; o0.w = acc[c][3] * inv;
        o1.x = acc[c][4] * inv; o1.y = acc[c][5] * inv;
        o1.z = acc[c][6] * inv; o1.w = acc[c][7] * inv;
        float* dst = orow + c * 16 + (lane >> 4) * 8;
        *(float4*)(dst)     = o0;
        *(float4*)(dst + 4) = o1;
    }
}

// ===========================================================================
extern "C" void kernel_launch(void* const* d_in, const int* in_sizes, int n_in,
                              void* d_out, int out_size, void* d_ws,
                              size_t ws_size, hipStream_t stream) {
    const float* x      = (const float*)d_in[0];
    const float* coords = (const float*)d_in[1];
    const float* mask   = (const float*)d_in[2];
    const float* qkv_w  = (const float*)d_in[3];
    const float* qkv_b  = (const float*)d_in[4];
    const float* rel_w  = (const float*)d_in[5];
    float* out = (float*)d_out;

    // workspace layout
    const size_t HEAD_ELEMS = (size_t)BATCH * NHEAD * SEQ * HD;   // 2,097,152
    const size_t X_ELEMS    = (size_t)BATCH * SEQ * D_MODEL;      // 2,097,152
    const size_t W_ELEMS    = (size_t)3 * D_MODEL * D_MODEL;      //   786,432
    unsigned short* wsq = (unsigned short*)d_ws;
    unsigned short* wsk = wsq + HEAD_ELEMS;
    unsigned short* wsv = wsk + HEAD_ELEMS;
    float*          cpw = (float*)(wsv + HEAD_ELEMS);
    unsigned short* xbw = (unsigned short*)(cpw + BATCH * NHEAD * SEQ);
    unsigned short* wbw = xbw + X_ELEMS;

    // 0) one-shot bf16 packing of x and qkv_w
    pack_bf16_kernel<<<(int)(X_ELEMS / 8 + 255) / 256, 256, 0, stream>>>(
        x, xbw, (int)(X_ELEMS / 8));
    pack_bf16_kernel<<<(int)(W_ELEMS / 8 + 255) / 256, 256, 0, stream>>>(
        qkv_w, wbw, (int)(W_ELEMS / 8));

    // 1) coordinate projection
    coord_proj_kernel<<<(BATCH * NHEAD * SEQ) / 256, 256, 0, stream>>>(
        coords, rel_w, cpw);

    // 2) QKV GEMM: grid = (1536/64, 4096/128)
    qkv_gemm_kernel<<<dim3(24, 32), 256, 0, stream>>>(
        xbw, wbw, qkv_b, wsq, wsk, wsv);

    // 3) flash attention: grid = (N/128, B*H)
    flash_attn_kernel<<<dim3(SEQ / 128, BATCH * NHEAD), 256, 0, stream>>>(
        wsq, wsk, wsv, cpw, mask, out);
}
